// HSTUMultiHeadAttention_45183055954603
// MI455X (gfx1250) — matrix operand order repaired and checked by reading, not compiled, not run on hardware
//
#include <hip/hip_runtime.h>
#include <math.h>

// ---------------------------------------------------------------------------
// CDNA5 (gfx1250) wave32 WMMA types / helpers
// ---------------------------------------------------------------------------
typedef __attribute__((ext_vector_type(16))) __bf16 v16bf;
typedef __attribute__((ext_vector_type(8)))  float  v8f;

union Frag {
  v16bf v;
  unsigned int u[8];
};

#define WMMA_BF16(a, b, c) \
  __builtin_amdgcn_wmma_f32_16x16x32_bf16(false, (a), false, (b), (short)0, (c), false, false)

__device__ __forceinline__ unsigned short f2bf(float f) {
  union { float f; unsigned u; } c; c.f = f;
  unsigned u = c.u + 0x7FFFu + ((c.u >> 16) & 1u);  // round-to-nearest-even
  return (unsigned short)(u >> 16);
}

// SiLU with hardware reciprocal (v_rcp_f32) instead of IEEE division chain.
__device__ __forceinline__ float fast_silu(float x) {
  return x * __builtin_amdgcn_rcpf(1.0f + __expf(-x));
}

__device__ __forceinline__ v8f vzero8() {
  v8f z;
#pragma unroll
  for (int i = 0; i < 8; ++i) z[i] = 0.0f;
  return z;
}

// Async global->LDS 16B copy (CDNA5, ASYNCcnt). Generic LDS pointer's low 32
// bits are the LDS offset (flat aperture rule), which is what VDST wants.
__device__ __forceinline__ void async_b128(void* lds, const void* g) {
  unsigned l = (unsigned)(uintptr_t)lds;
  asm volatile("global_load_async_to_lds_b128 %0, %1, off" :: "v"(l), "v"(g) : "memory");
}
__device__ __forceinline__ void wait_async() {
  asm volatile("s_wait_asynccnt 0" ::: "memory");
}

// A-matrix 16x32 bf16 fragment (ISA 7.12.2 table):
//   lane L: m = L%16, half h = L/16; VGPR v<4: K = 8h+2v ; v>=4: K = 16+8h+2(v-4)
__device__ __forceinline__ void load_a_frag(const unsigned short* lds, int row, int ld,
                                            int kBase, Frag& f) {
  const int lane = threadIdx.x & 31;
  const int h = lane >> 4;
  const unsigned short* p = lds + (row + (lane & 15)) * ld + kBase;
#pragma unroll
  for (int v = 0; v < 8; ++v) {
    const int k = (v < 4) ? (8 * h + 2 * v) : (16 + 8 * h + 2 * (v - 4));
    f.u[v] = *(const unsigned int*)(p + k);
  }
}

// B-matrix 32x16 bf16 fragment: lane L: n = L%16, half h = L/16; VGPR v<4: K = 8h+2v ; v>=4: K = 16+8h+2(v-4)
// LDS tile stored n-major, K-contiguous.
__device__ __forceinline__ void load_b_frag(const unsigned short* lds, int col, int ld,
                                            int kBase, Frag& f) {
  const int lane = threadIdx.x & 31;
  const int h = lane >> 4;
  const unsigned short* p = lds + (col + (lane & 15)) * ld + kBase;
#pragma unroll
  for (int v = 0; v < 8; ++v)
    f.u[v] = *(const unsigned int*)(p + ((v < 4) ? (8 * h + 2 * v) : (16 + 8 * h + 2 * (v - 4))));
}

// ---------------------------------------------------------------------------
// Kernel 0: affine LayerNorm of x -> bf16 normx   (8192 rows x 512)
// ---------------------------------------------------------------------------
__global__ __launch_bounds__(256) void k_ln(const float* __restrict__ x,
                                            const float* __restrict__ w,
                                            const float* __restrict__ b,
                                            unsigned short* __restrict__ out) {
  __shared__ float2 red[256];
  const int row = blockIdx.x;
  const int tid = threadIdx.x;
  const float* xr = x + (size_t)row * 512;
  const float a0 = xr[tid], a1 = xr[tid + 256];
  float2 acc; acc.x = a0 + a1; acc.y = a0 * a0 + a1 * a1;
  red[tid] = acc;
  __syncthreads();
  for (int s = 128; s > 0; s >>= 1) {
    if (tid < s) { red[tid].x += red[tid + s].x; red[tid].y += red[tid + s].y; }
    __syncthreads();
  }
  const float mu   = red[0].x * (1.0f / 512.0f);
  const float var  = red[0].y * (1.0f / 512.0f) - mu * mu;
  const float rstd = rsqrtf(var + 1e-6f);
  unsigned short* o = out + (size_t)row * 512;
  o[tid]       = f2bf((a0 - mu) * rstd * w[tid]       + b[tid]);
  o[tid + 256] = f2bf((a1 - mu) * rstd * w[tid + 256] + b[tid + 256]);
}

// ---------------------------------------------------------------------------
// f32 -> bf16 weight conversion
// ---------------------------------------------------------------------------
__global__ __launch_bounds__(256) void k_cvt(const float* __restrict__ in,
                                             unsigned short* __restrict__ out, int n) {
  const int i = blockIdx.x * 256 + threadIdx.x;
  if (i < n) out[i] = f2bf(in[i]);
}

// ---------------------------------------------------------------------------
// Kernel 1: projection GEMM  C[8192,2048] = normx[8192,512] @ uvqk[512,2048]
// SiLU epilogue, scatter into U(f32) / V / Q / K (bf16).
// Block tile 128x128, 8 waves, wave tile 32x64 (2x4 WMMA accumulators).
// A tile staged with async global->LDS (b128); B tile transposed via VGPRs.
// ---------------------------------------------------------------------------
#define LDK 40  // 32 + 8 pad (bf16): 80B rows -> 16B-aligned b128 LDS targets

__global__ __launch_bounds__(256) void k_proj(const unsigned short* __restrict__ A,
                                              const unsigned short* __restrict__ W,
                                              float* __restrict__ U,
                                              unsigned short* __restrict__ V,
                                              unsigned short* __restrict__ Qb,
                                              unsigned short* __restrict__ Kb) {
  __shared__ unsigned short As[128 * LDK];  // [m][k]
  __shared__ unsigned short Bs[128 * LDK];  // [n][k]  (transposed on store)
  const int tid  = threadIdx.x;
  const int lane = tid & 31, wave = tid >> 5;
  const int m0 = blockIdx.y * 128;
  const int n0 = blockIdx.x * 128;
  const int mw = (wave & 3) * 32;
  const int nw = (wave >> 2) * 64;
  v8f acc[2][4];
#pragma unroll
  for (int i = 0; i < 2; ++i)
#pragma unroll
    for (int j = 0; j < 4; ++j) acc[i][j] = vzero8();

  for (int k0 = 0; k0 < 512; k0 += 32) {
    __syncthreads();
    {  // A tile: 128 x 32, two async 16B chunks per thread
      const int r = tid >> 1, c = (tid & 1) * 16;
      const unsigned short* g = A + (m0 + r) * 512 + k0 + c;
      if (k0 + 32 < 512) __builtin_prefetch(g + 32, 0, 1);  // global_prefetch next tile
      async_b128(&As[r * LDK + c],     g);
      async_b128(&As[r * LDK + c + 8], g + 8);
    }
    {  // B tile transposed: W[k][n] -> Bs[n][k]
      const int k  = tid >> 3;
      const int ng = (tid & 7) * 16;
      const unsigned short* g = W + (k0 + k) * 2048 + n0 + ng;
#pragma unroll
      for (int j = 0; j < 16; ++j) Bs[(ng + j) * LDK + k] = g[j];
    }
    wait_async();
    __syncthreads();
    Frag af[2], bf[4];
#pragma unroll
    for (int i = 0; i < 2; ++i) load_a_frag(As, mw + 16 * i, LDK, 0, af[i]);
#pragma unroll
    for (int j = 0; j < 4; ++j) load_b_frag(Bs, nw + 16 * j, LDK, 0, bf[j]);
#pragma unroll
    for (int i = 0; i < 2; ++i)
#pragma unroll
      for (int j = 0; j < 4; ++j) acc[i][j] = WMMA_BF16(af[i].v, bf[j].v, acc[i][j]);
  }

  // SiLU + scatter. C layout: row = d + 8*(lane/16), col = lane%16.
  const int h = lane >> 4, nn = lane & 15;
#pragma unroll
  for (int i = 0; i < 2; ++i)
#pragma unroll
    for (int j = 0; j < 4; ++j)
#pragma unroll
      for (int d = 0; d < 8; ++d) {
        const float vv = fast_silu(acc[i][j][d]);
        const int m = m0 + mw + 16 * i + d + 8 * h;
        const int n = n0 + nw + 16 * j + nn;
        const int seg = n >> 9, c = n & 511;
        const size_t off = (size_t)m * 512 + c;
        if (seg == 0)      U[off]  = vv;
        else if (seg == 1) V[off]  = f2bf(vv);
        else if (seg == 2) Qb[off] = f2bf(vv);
        else               Kb[off] = f2bf(vv);
      }
}

// ---------------------------------------------------------------------------
// Kernel 2: causal HSTU attention per (b, head, 128-query tile).
// scores = silu(Q K^T)/2048 masked causal; out = scores @ V; per-row LN(64);
// gate with U; emit u_dot bf16. K tile staged with async global->LDS.
// ---------------------------------------------------------------------------
#define LD64 72  // 64 + 8 pad: 144B rows -> 16B-aligned b128 LDS targets

__global__ __launch_bounds__(256) void k_attn(const unsigned short* __restrict__ Qg,
                                              const unsigned short* __restrict__ Kg,
                                              const unsigned short* __restrict__ Vg,
                                              const float* __restrict__ U,
                                              unsigned short* __restrict__ ud) {
  __shared__ union {
    struct {
      unsigned short Ks[64 * LD64];      // [key][d]   (B-frag friendly for QK^T)
      unsigned short Vt[64 * LD64];      // [dv][key]  (B-frag friendly for P.V)
      unsigned short Ps[8][16 * LD64];   // per-wave bf16 scores restage
    } s;
    float Os[8][16][64];                 // per-wave f32 out staging (post-loop)
  } sh;

  const int tid = threadIdx.x, lane = tid & 31, wave = tid >> 5;
  const int bx = blockIdx.x;
  const int qt   = bx & 15;
  const int head = (bx >> 4) & 7;
  const int b    = bx >> 7;
  const int q0   = qt * 128;
  const int rowBase = b * 2048;
  const int hoff = head * 64;
  const int h = lane >> 4;

  // Q fragments straight from global (row-major, d contiguous -> A layout)
  const int qrow = q0 + wave * 16 + (lane & 15);
  const unsigned short* qgp = Qg + (size_t)(rowBase + qrow) * 512 + hoff;
  Frag qf[2];
#pragma unroll
  for (int ks = 0; ks < 2; ++ks)
#pragma unroll
    for (int v = 0; v < 8; ++v) {
      const int k = 32 * ks + ((v < 4) ? (8 * h + 2 * v) : (16 + 8 * h + 2 * (v - 4)));
      qf[ks].u[v] = *(const unsigned int*)(qgp + k);
    }

  v8f oacc[4];
#pragma unroll
  for (int j = 0; j < 4; ++j) oacc[j] = vzero8();

  const int jmax = 2 * qt + 1;               // causal: key block <= last query row
  for (int j = 0; j <= jmax; ++j) {
    __syncthreads();
    {  // K tile natural layout [key][d] via async b128 (2 chunks/thread)
      const int r = tid >> 2, c = (tid & 3) * 16;
      const unsigned short* g = Kg + (size_t)(rowBase + j * 64 + r) * 512 + hoff + c;
      async_b128(&sh.s.Ks[r * LD64 + c],     g);
      async_b128(&sh.s.Ks[r * LD64 + c + 8], g + 8);
    }
    {  // V tile transposed [dv][key] (scatter; cannot async-transpose)
      const int r = tid >> 2, c = (tid & 3) * 16;
      const unsigned short* g = Vg + (size_t)(rowBase + j * 64 + r) * 512 + hoff + c;
#pragma unroll
      for (int t = 0; t < 16; ++t) sh.s.Vt[(c + t) * LD64 + r] = g[t];
    }
    wait_async();
    __syncthreads();

    const int wq0 = q0 + wave * 16;
    if (j * 64 <= wq0 + 15) {                // skip fully-masked tiles
      v8f sacc[4];
#pragma unroll
      for (int jj = 0; jj < 4; ++jj) sacc[jj] = vzero8();
#pragma unroll
      for (int ks = 0; ks < 2; ++ks) {
        Frag kf;
#pragma unroll
        for (int jj = 0; jj < 4; ++jj) {
          load_b_frag(sh.s.Ks, jj * 16, LD64, 32 * ks, kf);
          sacc[jj] = WMMA_BF16(qf[ks].v, kf.v, sacc[jj]);
        }
      }
      // silu/S + causal mask, restage P as bf16 A-operand via own LDS slice
      unsigned short* P = sh.s.Ps[wave];
#pragma unroll
      for (int jj = 0; jj < 4; ++jj)
#pragma unroll
        for (int d = 0; d < 8; ++d) {
          float vv = fast_silu(sacc[jj][d]) * (1.0f / 2048.0f);
          const int qr = wq0 + d + 8 * h;
          const int kc = j * 64 + jj * 16 + (lane & 15);
          if (kc > qr) vv = 0.0f;
          P[(d + 8 * h) * LD64 + jj * 16 + (lane & 15)] = f2bf(vv);
        }
      asm volatile("s_wait_dscnt 0" ::: "memory");  // in-wave LDS store->load hazard
#pragma unroll
      for (int ks = 0; ks < 2; ++ks) {
        Frag pf, vf;
        load_a_frag(P, 0, LD64, 32 * ks, pf);
#pragma unroll
        for (int jj = 0; jj < 4; ++jj) {
          load_b_frag(sh.s.Vt, jj * 16, LD64, 32 * ks, vf);
          oacc[jj] = WMMA_BF16(pf.v, vf.v, oacc[jj]);
        }
      }
    }
  }

  __syncthreads();  // Os overlaps Ks/Vt/Ps in the union
#pragma unroll
  for (int jj = 0; jj < 4; ++jj)
#pragma unroll
    for (int d = 0; d < 8; ++d)
      sh.Os[wave][d + 8 * h][jj * 16 + (lane & 15)] = oacc[jj][d];
  __syncthreads();

  // per-row LN over 64 + U gating: 128 rows, one thread each
  if (tid < 128) {
    const int w2 = tid >> 4, rl = tid & 15;
    const float* row = sh.Os[w2][rl];
    float mu = 0.0f;
#pragma unroll
    for (int d = 0; d < 64; ++d) mu += row[d];
    mu *= (1.0f / 64.0f);
    float var = 0.0f;
#pragma unroll
    for (int d = 0; d < 64; ++d) { const float t = row[d] - mu; var += t * t; }
    var *= (1.0f / 64.0f);
    const float rstd = rsqrtf(var + 1e-6f);
    const int q = q0 + w2 * 16 + rl;
    const float* up = U + (size_t)(rowBase + q) * 512 + hoff;
    unsigned short* op = ud + (size_t)(rowBase + q) * 512 + hoff;
#pragma unroll
    for (int d = 0; d < 64; ++d)
      op[d] = f2bf((row[d] - mu) * rstd * up[d]);
  }
}

// ---------------------------------------------------------------------------
// Kernel 3: output GEMM  y[8192,512] = x + u_dot[8192,512] @ o_w[512,512] + o_b
// ---------------------------------------------------------------------------
__global__ __launch_bounds__(256) void k_out(const unsigned short* __restrict__ A,
                                             const unsigned short* __restrict__ W,
                                             const float* __restrict__ x,
                                             const float* __restrict__ ob,
                                             float* __restrict__ y) {
  __shared__ unsigned short As[128 * LDK];
  __shared__ unsigned short Bs[128 * LDK];
  const int tid  = threadIdx.x;
  const int lane = tid & 31, wave = tid >> 5;
  const int m0 = blockIdx.y * 128;
  const int n0 = blockIdx.x * 128;
  const int mw = (wave & 3) * 32;
  const int nw = (wave >> 2) * 64;
  v8f acc[2][4];
#pragma unroll
  for (int i = 0; i < 2; ++i)
#pragma unroll
    for (int j = 0; j < 4; ++j) acc[i][j] = vzero8();

  for (int k0 = 0; k0 < 512; k0 += 32) {
    __syncthreads();
    {
      const int r = tid >> 1, c = (tid & 1) * 16;
      const unsigned short* g = A + (m0 + r) * 512 + k0 + c;
      if (k0 + 32 < 512) __builtin_prefetch(g + 32, 0, 1);
      async_b128(&As[r * LDK + c],     g);
      async_b128(&As[r * LDK + c + 8], g + 8);
    }
    {
      const int k  = tid >> 3;
      const int ng = (tid & 7) * 16;
      const unsigned short* g = W + (k0 + k) * 512 + n0 + ng;
#pragma unroll
      for (int j = 0; j < 16; ++j) Bs[(ng + j) * LDK + k] = g[j];
    }
    wait_async();
    __syncthreads();
    Frag af[2], bf[4];
#pragma unroll
    for (int i = 0; i < 2; ++i) load_a_frag(As, mw + 16 * i, LDK, 0, af[i]);
#pragma unroll
    for (int j = 0; j < 4; ++j) load_b_frag(Bs, nw + 16 * j, LDK, 0, bf[j]);
#pragma unroll
    for (int i = 0; i < 2; ++i)
#pragma unroll
      for (int j = 0; j < 4; ++j) acc[i][j] = WMMA_BF16(af[i].v, bf[j].v, acc[i][j]);
  }

  const int h = lane >> 4, nn = lane & 15;
#pragma unroll
  for (int i = 0; i < 2; ++i)
#pragma unroll
    for (int j = 0; j < 4; ++j)
#pragma unroll
      for (int d = 0; d < 8; ++d) {
        const int m = m0 + mw + 16 * i + d + 8 * h;
        const int n = n0 + nw + 16 * j + nn;
        const size_t off = (size_t)m * 512 + n;
        y[off] = acc[i][j][d] + x[off] + ob[n];
      }
}

// ---------------------------------------------------------------------------
// Host launcher
// ---------------------------------------------------------------------------
extern "C" void kernel_launch(void* const* d_in, const int* in_sizes, int n_in,
                              void* d_out, int out_size, void* d_ws, size_t ws_size,
                              hipStream_t stream) {
  (void)in_sizes; (void)n_in; (void)out_size; (void)ws_size;
  const float* x    = (const float*)d_in[0];
  // d_in[1] = attn_mask (causal tril) -- never read; causality computed analytically
  const float* uvqk = (const float*)d_in[2];
  const float* o_w  = (const float*)d_in[3];
  const float* o_b  = (const float*)d_in[4];
  const float* ln_w = (const float*)d_in[5];
  const float* ln_b = (const float*)d_in[6];
  float* y = (float*)d_out;

  char* p = (char*)d_ws;
  auto alloc = [&](size_t bytes) { char* r = p; p += (bytes + 255) & ~(size_t)255; return r; };
  unsigned short* normx = (unsigned short*)alloc(8192ull * 512 * 2);
  unsigned short* Wb    = (unsigned short*)alloc(512ull * 2048 * 2);
  unsigned short* OWb   = (unsigned short*)alloc(512ull * 512 * 2);
  float*          Ub    = (float*)         alloc(8192ull * 512 * 4);
  unsigned short* Vb    = (unsigned short*)alloc(8192ull * 512 * 2);
  unsigned short* Qb    = (unsigned short*)alloc(8192ull * 512 * 2);
  unsigned short* Kb    = (unsigned short*)alloc(8192ull * 512 * 2);
  unsigned short* Ud    = (unsigned short*)alloc(8192ull * 512 * 2);

  k_ln  <<<8192, 256, 0, stream>>>(x, ln_w, ln_b, normx);
  k_cvt <<<(512 * 2048) / 256, 256, 0, stream>>>(uvqk, Wb, 512 * 2048);
  k_cvt <<<(512 * 512) / 256, 256, 0, stream>>>(o_w, OWb, 512 * 512);
  k_proj<<<dim3(16, 64), 256, 0, stream>>>(normx, Wb, Ub, Vb, Qb, Kb);
  k_attn<<<512, 256, 0, stream>>>(Qb, Kb, Vb, Ub, Ud);
  k_out <<<dim3(4, 64), 256, 0, stream>>>(Ud, OWb, x, o_b, y);
}
